// MSDeformAttn_11484742549687
// MI455X (gfx1250) — matrix operand order repaired and checked by reading, not compiled, not run on hardware
//
#include <hip/hip_runtime.h>
#include <hip/hip_bf16.h>

typedef __attribute__((ext_vector_type(16))) _Float16 v16h;
typedef __attribute__((ext_vector_type(8)))  _Float16 v8h;
typedef __attribute__((ext_vector_type(8)))  float    v8f;
typedef __attribute__((ext_vector_type(4)))  float    v4f;
typedef __attribute__((ext_vector_type(4)))  int      v4i;

#define NHEADS 8
#define NLEV   4
#define NPTS   4
#define DM     256
#define HD     32
#define NT_CAT 24    // 384/16 n-tiles (256 offset cols + 128 attn cols)
#define NT_OUT 16    // 256/16 n-tiles
#define KT     8     // 256/32 k-steps
#define BATCH_CONST 2
#define LDS_STRIDE 264  // padded row stride (528B) to spread LDS banks, 16B aligned

#if defined(__has_builtin)
#if __has_builtin(__builtin_amdgcn_global_load_async_to_lds_b128) && \
    __has_builtin(__builtin_amdgcn_s_wait_asynccnt)
#define HAVE_ASYNC_LDS 1
#endif
#endif
#ifndef HAVE_ASYNC_LDS
#define HAVE_ASYNC_LDS 0
#endif

// ---------------------------------------------------------------------------
// Pack f32 weights into WMMA B-fragment order:
//   packed[((nt*KT + kk)*32 + lane)*16 + j] = W[kk*32 + (lane<16?0:8) + (j<8?j:j+8)][nt*16 + (lane&15)]
// ---------------------------------------------------------------------------
__global__ void msda_pack_weights(const float* __restrict__ Woff,
                                  const float* __restrict__ Wattn,
                                  const float* __restrict__ Wout,
                                  _Float16* __restrict__ WcatP,
                                  _Float16* __restrict__ WoutP) {
    int t = blockIdx.x * blockDim.x + threadIdx.x;
    const int total_cat = NT_CAT * KT * 32;
    const int total_out = NT_OUT * KT * 32;
    if (t < total_cat) {
        int lane = t & 31;
        int kk   = (t >> 5) & 7;
        int nt   = t >> 8;
        int col  = nt * 16 + (lane & 15);
        int kb   = kk * 32 + ((lane & 16) ? 8 : 0);
        _Float16* dst = WcatP + (size_t)t * 16;
#pragma unroll
        for (int j = 0; j < 16; ++j) {
            int k = kb + (j < 8 ? j : j + 8);
            float v = (col < 256) ? Woff[(size_t)k * 256 + col]
                                  : Wattn[(size_t)k * 128 + (col - 256)];
            dst[j] = (_Float16)v;
        }
    } else if (t < total_cat + total_out) {
        int u    = t - total_cat;
        int lane = u & 31;
        int kk   = (u >> 5) & 7;
        int nt   = u >> 8;
        int col  = nt * 16 + (lane & 15);
        int kb   = kk * 32 + ((lane & 16) ? 8 : 0);
        _Float16* dst = WoutP + (size_t)u * 16;
#pragma unroll
        for (int j = 0; j < 16; ++j)
            dst[j] = (_Float16)Wout[(size_t)(kb + (j < 8 ? j : j + 8)) * 256 + col];
    }
}

// ---------------------------------------------------------------------------
// Feature map f32 -> f16 (halves the gather line size in the sampling stage).
// 8 elements per thread: 2x b128 load, 1x b128 store.
// ---------------------------------------------------------------------------
__global__ void msda_cvt_feat(const float* __restrict__ in,
                              _Float16* __restrict__ out, long long n8) {
    long long i = (long long)blockIdx.x * blockDim.x + threadIdx.x;
    if (i >= n8) return;
    const float* s = in + i * 8;
    v4f a = *(const v4f*)s;
    v4f b = *(const v4f*)(s + 4);
    v8h o;
#pragma unroll
    for (int j = 0; j < 4; ++j) { o[j] = (_Float16)a[j]; o[4 + j] = (_Float16)b[j]; }
    *(v8h*)(out + i * 8) = o;
}

// ---------------------------------------------------------------------------
// Projection GEMM: [M,256] x [256,384] -> offsets f16 [M,256] + attn logits f32 [M,128]
// One block = one 16-row M tile, 8 waves x 3 N-tiles each.
// Query tile staged to LDS (async-to-LDS path when available), converted to f16.
// ---------------------------------------------------------------------------
__global__ __launch_bounds__(256) void msda_proj_gemm(
    const float*    __restrict__ query,
    const _Float16* __restrict__ WcatP,
    const float*    __restrict__ b_off,
    const float*    __restrict__ b_attn,
    _Float16*       __restrict__ off_buf,
    float*          __restrict__ attn_buf,
    int M)
{
    __shared__ _Float16 As[16 * LDS_STRIDE];
    const int mt  = blockIdx.x;
    const int tid = threadIdx.x;

#if HAVE_ASYNC_LDS
    __shared__ float Qstage[16 * DM];
    {
        // 16 rows x 256 f32 = 1024 x 16B chunks; 256 threads x 4 async copies.
#pragma unroll
        for (int i = 0; i < 4; ++i) {
            int e = (i << 8) | tid;
            int r = e >> 6;
            int c = (e & 63) << 2;
            int row = mt * 16 + r;
            if (row >= M) row = M - 1;
            __builtin_amdgcn_global_load_async_to_lds_b128(
                (v4i*)(query + (size_t)row * DM + c),
                (v4i*)(Qstage + r * DM + c),
                0, 0);
        }
        __builtin_amdgcn_s_wait_asynccnt(0);
        __syncthreads();
        int r  = tid >> 4;
        int c0 = (tid & 15) * 16;
        const float* src = Qstage + r * DM + c0;
        _Float16*    dst = As + r * LDS_STRIDE + c0;
#pragma unroll
        for (int j = 0; j < 16; ++j) dst[j] = (_Float16)src[j];
    }
#else
    {
        int r   = tid >> 4;
        int c0  = (tid & 15) * 16;
        int row = mt * 16 + r;
        if (row >= M) row = M - 1;
        const float* src = query + (size_t)row * DM + c0;
        _Float16*    dst = As + r * LDS_STRIDE + c0;
#pragma unroll
        for (int j = 0; j < 16; ++j) dst[j] = (_Float16)src[j];
    }
#endif
    __syncthreads();

    const int wave = tid >> 5;
    const int lane = tid & 31;
    const int rowA = lane & 15;
    const int hi   = (lane >> 4) & 1;
    const bool full = (mt * 16 + 16) <= M;

    __builtin_prefetch(WcatP + (((size_t)wave * KT) * 32 + lane) * 16, 0, 1);

    // A fragments (ISA 16-bit A layout: lane = M, half-wave selects K+8 / K+24).
    v16h afrag[KT];
#pragma unroll
    for (int kk = 0; kk < KT; ++kk) {
        const _Float16* base = As + rowA * LDS_STRIDE + kk * 32 + hi * 8;
        v8h lo = *(const v8h*)(base);
        v8h h8 = *(const v8h*)(base + 16);
        v16h a;
#pragma unroll
        for (int j = 0; j < 8; ++j) { a[j] = lo[j]; a[8 + j] = h8[j]; }
        afrag[kk] = a;
    }

#pragma unroll
    for (int it = 0; it < 3; ++it) {
        const int nt = wave + it * 8;        // 0..23; it<2 -> offset cols, it==2 -> attn cols
        v8f acc = {};
#pragma unroll
        for (int kk = 0; kk < KT; ++kk) {
            const _Float16* bp = WcatP + (((size_t)nt * KT + kk) * 32 + lane) * 16;
            v16h b = *(const v16h*)bp;
            acc = __builtin_amdgcn_wmma_f32_16x16x32_f16(
                      false, afrag[kk], false, b, (short)0, acc, false, false);
        }
        const int colg = nt * 16 + (lane & 15);
        if (it < 2) {                        // offset outputs -> f16
            float bias = b_off[colg];
            if (full) {
#pragma unroll
                for (int r = 0; r < 8; ++r) {
                    int row = mt * 16 + r + hi * 8;
                    off_buf[(size_t)row * 256 + colg] = (_Float16)(acc[r] + bias);
                }
            } else {
#pragma unroll
                for (int r = 0; r < 8; ++r) {
                    int row = mt * 16 + r + hi * 8;
                    if (row < M)
                        off_buf[(size_t)row * 256 + colg] = (_Float16)(acc[r] + bias);
                }
            }
        } else {                             // attn logits -> f32
            float bias = b_attn[colg - 256];
            if (full) {
#pragma unroll
                for (int r = 0; r < 8; ++r) {
                    int row = mt * 16 + r + hi * 8;
                    attn_buf[(size_t)row * 128 + (colg - 256)] = acc[r] + bias;
                }
            } else {
#pragma unroll
                for (int r = 0; r < 8; ++r) {
                    int row = mt * 16 + r + hi * 8;
                    if (row < M)
                        attn_buf[(size_t)row * 128 + (colg - 256)] = acc[r] + bias;
                }
            }
        }
    }
}

// ---------------------------------------------------------------------------
// Per-(query,head) softmax over 16 (level x point) logits, in place.
// ---------------------------------------------------------------------------
__global__ void msda_softmax16(float* __restrict__ attn_buf, int M) {
    int t = blockIdx.x * blockDim.x + threadIdx.x;
    if (t >= M * NHEADS) return;
    float* p = attn_buf + (size_t)(t >> 3) * 128 + (t & 7) * 16;
    float mx = p[0];
#pragma unroll
    for (int i = 1; i < 16; ++i) mx = fmaxf(mx, p[i]);
    float e[16];
    float s = 0.f;
#pragma unroll
    for (int i = 0; i < 16; ++i) { e[i] = __expf(p[i] - mx); s += e[i]; }
    float inv = 1.f / s;
#pragma unroll
    for (int i = 0; i < 16; ++i) p[i] = e[i] * inv;
}

// ---------------------------------------------------------------------------
// Bilinear sampling + weighted accumulation over the f16 feature map.
// One block = one query, one wave = one head, lane = head-dim channel.
// Corner gathers are contiguous 64B lines (32 halves) -> fully coalesced.
// ---------------------------------------------------------------------------
__global__ __launch_bounds__(256) void msda_sample(
    const _Float16*  __restrict__ feat_h,
    const float*     __restrict__ ref_pts,
    const long long* __restrict__ spatial_shapes,
    const long long* __restrict__ level_start,
    const _Float16*  __restrict__ off_buf,
    const float*     __restrict__ attn_buf,
    _Float16*        __restrict__ val_buf,
    int LQ, int LEN)
{
    const int m    = blockIdx.x;           // b*LQ + q
    const int h    = threadIdx.x >> 5;     // head
    const int lane = threadIdx.x & 31;     // head-dim channel
    const int b    = m / LQ;

    const float rx = ref_pts[2 * m + 0];
    const float ry = ref_pts[2 * m + 1];
    const _Float16* feat = feat_h + (size_t)b * LEN * DM + h * HD + lane;
    const _Float16* offr = off_buf  + (size_t)m * 256;
    const float*    attr = attn_buf + (size_t)m * 128;

    float acc = 0.f;
#pragma unroll
    for (int l = 0; l < NLEV; ++l) {
        const int W  = (int)spatial_shapes[2 * l + 1];
        const int H  = (int)spatial_shapes[2 * l + 0];
        const int st = (int)level_start[l];
        const _Float16* fl = feat + (size_t)st * DM;
        const float invW = 1.f / (float)W;
        const float invH = 1.f / (float)H;
#pragma unroll
        for (int p = 0; p < NPTS; ++p) {
            int   oidx = ((h * NLEV + l) * NPTS + p) * 2;
            float lx   = rx + (float)offr[oidx + 0] * invW;   // norm = (W,H)
            float ly   = ry + (float)offr[oidx + 1] * invH;
            float aw   = attr[h * 16 + l * NPTS + p];
            float x = lx * (float)W - 0.5f;
            float y = ly * (float)H - 0.5f;
            float x0 = floorf(x), y0 = floorf(y);
            float wx1 = x - x0, wy1 = y - y0;
            float wx0 = 1.f - wx1, wy0 = 1.f - wy1;
            int x0i = (int)x0, y0i = (int)y0;
#pragma unroll
            for (int c = 0; c < 4; ++c) {
                int dx = c & 1, dy = c >> 1;
                int xi = x0i + dx, yi = y0i + dy;
                float cw = (dx ? wx1 : wx0) * (dy ? wy1 : wy0);
                if (xi >= 0 && xi < W && yi >= 0 && yi < H) {
                    acc += aw * cw * (float)fl[(size_t)(yi * W + xi) * DM];
                }
            }
        }
    }
    val_buf[(size_t)m * DM + h * HD + lane] = (_Float16)acc;
}

// ---------------------------------------------------------------------------
// Output GEMM: val [M,256] f16 x W_out [256,256] -> out [M,256] f32 (+bias).
// ---------------------------------------------------------------------------
__global__ __launch_bounds__(256) void msda_out_gemm(
    const _Float16* __restrict__ val_buf,
    const _Float16* __restrict__ WoutP,
    const float*    __restrict__ b_out,
    float*          __restrict__ out,
    int M)
{
    const int mt   = blockIdx.x;
    const int tid  = threadIdx.x;
    const int wave = tid >> 5;
    const int lane = tid & 31;
    const int rowA = lane & 15;
    const int hi   = (lane >> 4) & 1;
    const bool full = (mt * 16 + 16) <= M;

    int row = mt * 16 + rowA;
    if (row >= M) row = M - 1;
    const _Float16* arow = val_buf + (size_t)row * DM;

    __builtin_prefetch(WoutP + (((size_t)wave * KT) * 32 + lane) * 16, 0, 1);

    v16h afrag[KT];
#pragma unroll
    for (int kk = 0; kk < KT; ++kk) {
        const _Float16* base = arow + kk * 32 + hi * 8;
        v8h lo = *(const v8h*)(base);
        v8h h8 = *(const v8h*)(base + 16);
        v16h a;
#pragma unroll
        for (int j = 0; j < 8; ++j) { a[j] = lo[j]; a[8 + j] = h8[j]; }
        afrag[kk] = a;
    }

#pragma unroll
    for (int it = 0; it < 2; ++it) {
        int nt = wave + it * 8;            // 0..15
        v8f acc = {};
#pragma unroll
        for (int kk = 0; kk < KT; ++kk) {
            const _Float16* bp = WoutP + (((size_t)nt * KT + kk) * 32 + lane) * 16;
            v16h b = *(const v16h*)bp;
            acc = __builtin_amdgcn_wmma_f32_16x16x32_f16(
                      false, afrag[kk], false, b, (short)0, acc, false, false);
        }
        int colg = nt * 16 + (lane & 15);
        float bias = b_out[colg];
        if (full) {
#pragma unroll
            for (int r = 0; r < 8; ++r) {
                int rg = mt * 16 + r + hi * 8;
                out[(size_t)rg * DM + colg] = acc[r] + bias;
            }
        } else {
#pragma unroll
            for (int r = 0; r < 8; ++r) {
                int rg = mt * 16 + r + hi * 8;
                if (rg < M) out[(size_t)rg * DM + colg] = acc[r] + bias;
            }
        }
    }
}

// ---------------------------------------------------------------------------
extern "C" void kernel_launch(void* const* d_in, const int* in_sizes, int n_in,
                              void* d_out, int out_size, void* d_ws, size_t ws_size,
                              hipStream_t stream) {
    (void)n_in; (void)out_size; (void)ws_size;

    const float*     query = (const float*)d_in[0];
    const float*     refp  = (const float*)d_in[1];
    const float*     inpf  = (const float*)d_in[2];
    const long long* ss    = (const long long*)d_in[3];
    const long long* lsi   = (const long long*)d_in[4];
    const float*     Woff  = (const float*)d_in[5];
    const float*     boff  = (const float*)d_in[6];
    const float*     Wattn = (const float*)d_in[7];
    const float*     battn = (const float*)d_in[8];
    const float*     Wout  = (const float*)d_in[9];
    const float*     bout  = (const float*)d_in[10];
    float* out = (float*)d_out;

    const int M   = in_sizes[0] / DM;        // B * LQ
    const int LQ  = M / BATCH_CONST;
    const int LEN = in_sizes[2] / DM / BATCH_CONST;

    // ---- workspace carve (all offsets 256B-aligned) ----
    char* ws = (char*)d_ws;
    const size_t cat_bytes  = (size_t)NT_CAT * KT * 32 * 16 * sizeof(_Float16);  // 196608
    const size_t wout_bytes = (size_t)NT_OUT * KT * 32 * 16 * sizeof(_Float16);  // 131072
    size_t o = 0;
    _Float16* WcatP    = (_Float16*)(ws + o); o += cat_bytes;
    _Float16* WoutP    = (_Float16*)(ws + o); o += wout_bytes;
    _Float16* off_buf  = (_Float16*)(ws + o); o += (size_t)M * 256 * sizeof(_Float16);
    float*    attn_buf = (float*)   (ws + o); o += (size_t)M * 128 * sizeof(float);
    _Float16* val_buf  = (_Float16*)(ws + o); o += (size_t)M * 256 * sizeof(_Float16);
    _Float16* feat_h   = (_Float16*)(ws + o); o += (size_t)M * 256 * sizeof(_Float16);

    const int mtiles = (M + 15) / 16;

    // 1. pack weights into WMMA B-fragment order
    {
        int total = (NT_CAT + NT_OUT) * KT * 32;     // 10240
        msda_pack_weights<<<(total + 255) / 256, 256, 0, stream>>>(
            Woff, Wattn, Wout, WcatP, WoutP);
    }
    // 2. feature map f32 -> f16 (halves gather traffic in stage 5)
    {
        long long n8 = (long long)M * (DM / 8);
        msda_cvt_feat<<<(int)((n8 + 255) / 256), 256, 0, stream>>>(inpf, feat_h, n8);
    }
    // 3. offset + attn-logit projection (WMMA)
    msda_proj_gemm<<<mtiles, 256, 0, stream>>>(
        query, WcatP, boff, battn, off_buf, attn_buf, M);
    // 4. softmax over the 16 (level,point) logits per head
    msda_softmax16<<<(M * NHEADS + 255) / 256, 256, 0, stream>>>(attn_buf, M);
    // 5. bilinear sampling + attention-weighted accumulation
    msda_sample<<<M, 256, 0, stream>>>(
        feat_h, refp, ss, lsi, off_buf, attn_buf, val_buf, LQ, LEN);
    // 6. output projection (WMMA) + bias
    msda_out_gemm<<<mtiles, 256, 0, stream>>>(val_buf, WoutP, bout, out, M);
}
